// TransformerEncoderLayer_t5_919123001864
// MI455X (gfx1250) — matrix operand order repaired and checked by reading, not compiled, not run on hardware
//
#include <hip/hip_runtime.h>
#include <hip/hip_bf16.h>

#define SEQL   2048
#define DMODEL 512
#define DFF    2048
#define NHEADS 8
#define DHEAD  64
#define BATCHN 4

typedef __attribute__((ext_vector_type(16))) _Float16 v16h;
typedef __attribute__((ext_vector_type(8)))  _Float16 v8h;
typedef __attribute__((ext_vector_type(4)))  _Float16 v4h;
typedef __attribute__((ext_vector_type(8)))  float    v8f;

__device__ __forceinline__ v8f wmma_f16(v16h a, v16h b, v8f c) {
  // D = A(16x32) * B(32x16) + C, f32 accumulate
  return __builtin_amdgcn_wmma_f32_16x16x32_f16(false, a, false, b, (short)0, c, false, false);
}

__device__ __forceinline__ v16h cat8(v8h lo, v8h hi) {
  return __builtin_shufflevector(lo, hi, 0, 1, 2, 3, 4, 5, 6, 7,
                                 8, 9, 10, 11, 12, 13, 14, 15);
}

// ---- fragment loaders (layouts per CDNA5 ISA §7.12.2) -----------
// All LDS row strides are multiples of 8 halfs (16 B) so the v8h loads below
// are 16-byte aligned and lower to ds_load_b128.

// A fragment: 16(M) x 32(K) from LDS row-major tile [row][k], ld in halfs.
// lanes 0-15: M=lane, VGPR0-3 -> K=0..7, VGPR4-7 -> K=16..23
// lanes 16-31: M=lane-16, VGPR0-3 -> K=8..15, VGPR4-7 -> K=24..31
__device__ __forceinline__ v16h load_a_frag(const _Float16* __restrict__ base, int ld,
                                            int row0, int k0) {
  const int lane = threadIdx.x & 31;
  const _Float16* r = base + (row0 + (lane & 15)) * ld + k0 + (lane >> 4) * 8;
  v8h lo = *(const v8h*)r;          // K = k0 + h*8 + [0,8)
  v8h hi = *(const v8h*)(r + 16);   // K = k0 + 16 + h*8 + [0,8)
  return cat8(lo, hi);
}

// B fragment: 32(K) x 16(N) where LDS storage is [n][k] (K contiguous per col).
// lane = N (&15); lanes 0-15 hold K=0..7,16..23, lanes 16-31 K=8..15,24..31.
__device__ __forceinline__ v16h load_b_frag(const _Float16* __restrict__ base, int ld,
                                            int row0, int k0) {
  const int lane = threadIdx.x & 31;
  const _Float16* r = base + (row0 + (lane & 15)) * ld + k0 + (lane >> 4) * 8;
  v8h lo = *(const v8h*)r;          // K = 8h + [0,8)
  v8h hi = *(const v8h*)(r + 16);   // K = 16 + 8h + [0,8)
  return cat8(lo, hi);
}

// ---- global load helpers (always into float[4] regs) ------------------------
__device__ __forceinline__ void gload4(float* dst, const float* src) {
  float4 t = *(const float4*)src;   // global_load_b128
  dst[0] = t.x; dst[1] = t.y; dst[2] = t.z; dst[3] = t.w;
}
__device__ __forceinline__ void gload4(float* dst, const _Float16* src) {
  v4h t = *(const v4h*)src;         // global_load_b64
  dst[0] = (float)t[0]; dst[1] = (float)t[1];
  dst[2] = (float)t[2]; dst[3] = (float)t[3];
}

// T5 relative-position bucket (num_buckets=32 -> bidirectional halves of 16,
// max_exact=8, max_distance=128; log(128/8)=log(16)=2.7725887)
__device__ __forceinline__ int rel_bucket(int n /* q - k */) {
  int ret = 0;
  if (n < 0) { ret = 16; n = -n; }
  int b;
  if (n < 8) {
    b = n;
  } else {
    float f = __logf((float)n * 0.125f) * (8.0f / 2.7725887f);
    b = 8 + (int)f;
    if (b > 15) b = 15;
  }
  return ret + b;
}

// ---- GEMM: C[M,N] = act(A[M,K] @ W[K,N] + bias[N]) --------------------------
// Block tile 128x64, 8 waves (4 row-groups x 2 col-groups), 32x32 per wave.
// Global tiles double-buffered through registers to overlap VMEM with WMMA.
template <typename AT, typename CT, bool RELU>
__global__ __launch_bounds__(256) void wmma_gemm(const AT* __restrict__ A,
                                                 const float* __restrict__ W,
                                                 const float* __restrict__ bias,
                                                 CT* __restrict__ C,
                                                 int M, int N, int K) {
  __shared__ _Float16 As[128][40];   // [m][k]
  __shared__ _Float16 Bst[64][40];   // [n][k]  (transposed on stage)

  const int tid  = threadIdx.x;
  const int wave = tid >> 5;
  const int wr   = wave >> 1;   // 0..3
  const int wc   = wave & 1;    // 0..1
  const int lane = tid & 31;
  const int lh   = lane >> 4;
  const int ln   = lane & 15;
  const int m0   = blockIdx.y * 128;
  const int n0   = blockIdx.x * 64;

  // staging coordinates (constant over K loop)
  int arr[4], acc_[4], brr[2], bcc[2];
#pragma unroll
  for (int it = 0; it < 4; ++it) {
    int gid = tid + 256 * it;
    arr[it] = gid >> 3;          // 0..127
    acc_[it] = (gid & 7) * 4;    // 0..28
  }
#pragma unroll
  for (int it = 0; it < 2; ++it) {
    int gid = tid + 256 * it;
    brr[it] = gid >> 4;          // 0..31  (k within tile)
    bcc[it] = (gid & 15) * 4;    // 0..60  (n within tile)
  }

  v8f acc[2][2];
#pragma unroll
  for (int i = 0; i < 2; ++i)
#pragma unroll
    for (int j = 0; j < 2; ++j)
      acc[i][j] = {};

  // prefetch first K-tile into registers
  float ra[4][4], rb[2][4];
#pragma unroll
  for (int it = 0; it < 4; ++it)
    gload4(ra[it], A + (size_t)(m0 + arr[it]) * K + acc_[it]);
#pragma unroll
  for (int it = 0; it < 2; ++it)
    gload4(rb[it], W + (size_t)brr[it] * N + (n0 + bcc[it]));

  for (int k0 = 0; k0 < K; k0 += 32) {
    __syncthreads();
    // stage registers -> LDS (B transposed)
#pragma unroll
    for (int it = 0; it < 4; ++it)
#pragma unroll
      for (int i = 0; i < 4; ++i)
        As[arr[it]][acc_[it] + i] = (_Float16)ra[it][i];
#pragma unroll
    for (int it = 0; it < 2; ++it)
#pragma unroll
      for (int i = 0; i < 4; ++i)
        Bst[bcc[it] + i][brr[it]] = (_Float16)rb[it][i];
    __syncthreads();

    // issue next tile's global loads (overlaps with WMMA below)
    if (k0 + 32 < K) {
#pragma unroll
      for (int it = 0; it < 4; ++it)
        gload4(ra[it], A + (size_t)(m0 + arr[it]) * K + (k0 + 32 + acc_[it]));
#pragma unroll
      for (int it = 0; it < 2; ++it)
        gload4(rb[it], W + (size_t)(k0 + 32 + brr[it]) * N + (n0 + bcc[it]));
    }

    v16h bf[2];
#pragma unroll
    for (int j = 0; j < 2; ++j)
      bf[j] = load_b_frag(&Bst[0][0], 40, wc * 32 + j * 16, 0);
#pragma unroll
    for (int i = 0; i < 2; ++i) {
      v16h a = load_a_frag(&As[0][0], 40, wr * 32 + i * 16, 0);
#pragma unroll
      for (int j = 0; j < 2; ++j)
        acc[i][j] = wmma_f16(a, bf[j], acc[i][j]);
    }
  }

  // epilogue: C layout VGPR e -> M = e + 8*half, N = lane&15
#pragma unroll
  for (int i = 0; i < 2; ++i) {
#pragma unroll
    for (int j = 0; j < 2; ++j) {
#pragma unroll
      for (int e = 0; e < 8; ++e) {
        int m = m0 + wr * 32 + i * 16 + e + 8 * lh;
        int n = n0 + wc * 32 + j * 16 + ln;
        float v = acc[i][j][e] + bias[n];
        if (RELU) v = fmaxf(v, 0.0f);
        C[(size_t)m * N + n] = (CT)v;
      }
    }
  }
}

// ---- Flash attention with T5 relative bias ---------------------------------
// One block per (batch, head, 64-query tile). K/V streamed in 32-key tiles,
// double-buffered through registers. Output accumulator lives in wave VGPRs;
// the online-softmax rescale is folded into the WMMA C operand via alpha[row].
__global__ __launch_bounds__(256) void flash_attn(const float* __restrict__ Q,
                                                  const float* __restrict__ K,
                                                  const float* __restrict__ V,
                                                  const float* __restrict__ rel_bias,
                                                  float* __restrict__ CTX) {
  __shared__ _Float16 Qs[64][72];   // [q][d]
  __shared__ _Float16 Ks[32][72];   // [key][d]   (B-frag for Q@K^T reads rows)
  __shared__ _Float16 Vst[64][40];  // [d][key]   (transposed on stage)
  __shared__ _Float16 Ps[64][40];   // [q][key]
  __shared__ float Ss[64][36];
  __shared__ float rowm[64];
  __shared__ float rowl[64];
  __shared__ float rowa[64];        // per-row alpha for this k-tile
  __shared__ float biasTab[32];

  const int tid  = threadIdx.x;
  const int wave = tid >> 5;
  const int wr   = wave >> 1;  // 0..3 -> q rows [wr*16, +16)
  const int wc   = wave & 1;   // 0..1
  const int lane = tid & 31;
  const int lh   = lane >> 4;
  const int ln   = lane & 15;

  const int q0 = blockIdx.x * 64;
  const int h  = blockIdx.y;
  const int b  = blockIdx.z;
  const size_t base = (size_t)b * SEQL * DMODEL + (size_t)h * DHEAD;

  // staging coordinates for K/V tiles (32 x 64)
  int krr[2], kcc[2];
#pragma unroll
  for (int it = 0; it < 2; ++it) {
    int gid = tid + 256 * it;
    krr[it] = gid >> 4;          // key row 0..31
    kcc[it] = (gid & 15) * 4;    // d col  0..60
  }

  // stage Q tile 64x64
#pragma unroll
  for (int it = 0; it < 4; ++it) {
    int gid = tid + 256 * it;
    int r = gid >> 4;
    int c = (gid & 15) * 4;
    float rq[4];
    gload4(rq, Q + base + (size_t)(q0 + r) * DMODEL + c);
#pragma unroll
    for (int i = 0; i < 4; ++i) Qs[r][c + i] = (_Float16)rq[i];
  }
  if (tid < 64) { rowm[tid] = -1e30f; rowl[tid] = 0.0f; }
  if (tid < 32) biasTab[tid] = rel_bias[h * 32 + tid];

  // O accumulator in registers: wave owns rows [wr*16,+16), d-cols wc*32+ct*16
  v8f oac[2];
  oac[0] = {};
  oac[1] = {};

  // prefetch first K/V tile
  float rk[2][4], rv[2][4];
#pragma unroll
  for (int it = 0; it < 2; ++it) {
    gload4(rk[it], K + base + (size_t)krr[it] * DMODEL + kcc[it]);
    gload4(rv[it], V + base + (size_t)krr[it] * DMODEL + kcc[it]);
  }

  for (int k0 = 0; k0 < SEQL; k0 += 32) {
    __syncthreads();
    // stage K (row-major) and V (transposed) tiles
#pragma unroll
    for (int it = 0; it < 2; ++it) {
#pragma unroll
      for (int i = 0; i < 4; ++i) {
        Ks[krr[it]][kcc[it] + i] = (_Float16)rk[it][i];
        Vst[kcc[it] + i][krr[it]] = (_Float16)rv[it][i];
      }
    }
    __syncthreads();

    // issue next tile's global loads (overlap with compute below)
    if (k0 + 32 < SEQL) {
#pragma unroll
      for (int it = 0; it < 2; ++it) {
        gload4(rk[it], K + base + (size_t)(k0 + 32 + krr[it]) * DMODEL + kcc[it]);
        gload4(rv[it], V + base + (size_t)(k0 + 32 + krr[it]) * DMODEL + kcc[it]);
      }
    }

    // scores: wave (wr,wc) computes 16x16 tile; K-dim = d_head = 64 -> 2 WMMAs
    {
      v8f sc = {};
#pragma unroll
      for (int ks = 0; ks < 64; ks += 32) {
        v16h a  = load_a_frag(&Qs[0][0], 72, wr * 16, ks);
        v16h bt = load_b_frag(&Ks[0][0], 72, wc * 16, ks);
        sc = wmma_f16(a, bt, sc);
      }
#pragma unroll
      for (int e = 0; e < 8; ++e) {
        int ql = wr * 16 + e + 8 * lh;
        int kl = wc * 16 + ln;
        int bkt = rel_bucket((q0 + ql) - (k0 + kl));
        Ss[ql][kl] = (sc[e] + biasTab[bkt]) * 0.125f;  // /sqrt(64)
      }
    }
    __syncthreads();

    // online softmax update: one thread per query row
    if (tid < 64) {
      const int r = tid;
      float mold = rowm[r];
      float mnew = mold;
#pragma unroll
      for (int c = 0; c < 32; ++c) mnew = fmaxf(mnew, Ss[r][c]);
      float alpha = __expf(mold - mnew);
      float lsum = 0.0f;
#pragma unroll
      for (int c = 0; c < 32; ++c) {
        float p = __expf(Ss[r][c] - mnew);
        Ps[r][c] = (_Float16)p;
        lsum += p;
      }
      rowl[r] = rowl[r] * alpha + lsum;
      rowm[r] = mnew;
      rowa[r] = alpha;
    }
    __syncthreads();

    // P @ V with O-rescale folded into the WMMA C operand:
    //   oac = P @ V + alpha[row] * oac
    {
      float al[8];
#pragma unroll
      for (int e = 0; e < 8; ++e) al[e] = rowa[wr * 16 + e + 8 * lh];
      v16h a = load_a_frag(&Ps[0][0], 40, wr * 16, 0);
#pragma unroll
      for (int ct = 0; ct < 2; ++ct) {
        v16h bv = load_b_frag(&Vst[0][0], 40, wc * 32 + ct * 16, 0);
        v8f cin;
#pragma unroll
        for (int e = 0; e < 8; ++e) cin[e] = oac[ct][e] * al[e];
        oac[ct] = wmma_f16(a, bv, cin);
      }
    }
  }
  __syncthreads();

  // normalize and write context (layout [B, S, H*DHEAD] = [B, S, DMODEL])
  {
    float linv[8];
#pragma unroll
    for (int e = 0; e < 8; ++e)
      linv[e] = 1.0f / rowl[wr * 16 + e + 8 * lh];
#pragma unroll
    for (int ct = 0; ct < 2; ++ct) {
#pragma unroll
      for (int e = 0; e < 8; ++e) {
        int row = wr * 16 + e + 8 * lh;
        int col = wc * 32 + ct * 16 + ln;
        CTX[base + (size_t)(q0 + row) * DMODEL + col] = oac[ct][e] * linv[e];
      }
    }
  }
}

// ---- fused residual-add + layernorm (one block per row of 512) -------------
__global__ __launch_bounds__(256) void add_ln(const float* __restrict__ X,
                                              const float* __restrict__ Y,
                                              const float* __restrict__ scale,
                                              const float* __restrict__ bias,
                                              float* __restrict__ Out) {
  __shared__ float red[256];
  const int row = blockIdx.x;
  const int t = threadIdx.x;
  const float* xr = X + (size_t)row * DMODEL;
  const float* yr = Y + (size_t)row * DMODEL;
  float v0 = xr[t] + yr[t];
  float v1 = xr[t + 256] + yr[t + 256];
  red[t] = v0 + v1;
  __syncthreads();
  for (int off = 128; off > 0; off >>= 1) {
    if (t < off) red[t] += red[t + off];
    __syncthreads();
  }
  float mu = red[0] * (1.0f / 512.0f);
  __syncthreads();
  float d0 = v0 - mu, d1 = v1 - mu;
  red[t] = d0 * d0 + d1 * d1;
  __syncthreads();
  for (int off = 128; off > 0; off >>= 1) {
    if (t < off) red[t] += red[t + off];
    __syncthreads();
  }
  float rs = rsqrtf(red[0] * (1.0f / 512.0f) + 1e-6f);
  Out[(size_t)row * DMODEL + t]       = d0 * rs * scale[t] + bias[t];
  Out[(size_t)row * DMODEL + t + 256] = d1 * rs * scale[t + 256] + bias[t + 256];
}

// ---- host orchestration -----------------------------------------------------
extern "C" void kernel_launch(void* const* d_in, const int* in_sizes, int n_in,
                              void* d_out, int out_size, void* d_ws, size_t ws_size,
                              hipStream_t stream) {
  (void)in_sizes; (void)n_in; (void)out_size; (void)ws_size;
  const float* x   = (const float*)d_in[0];
  const float* Wq  = (const float*)d_in[1];
  const float* bq  = (const float*)d_in[2];
  const float* Wk  = (const float*)d_in[3];
  const float* bk  = (const float*)d_in[4];
  const float* Wv  = (const float*)d_in[5];
  const float* bv  = (const float*)d_in[6];
  const float* Wo  = (const float*)d_in[7];
  const float* bo  = (const float*)d_in[8];
  const float* rb  = (const float*)d_in[9];
  const float* W1  = (const float*)d_in[10];
  const float* b1  = (const float*)d_in[11];
  const float* W2  = (const float*)d_in[12];
  const float* b2  = (const float*)d_in[13];
  const float* l1s = (const float*)d_in[14];
  const float* l1b = (const float*)d_in[15];
  const float* l2s = (const float*)d_in[16];
  const float* l2b = (const float*)d_in[17];

  char* ws = (char*)d_ws;
  const size_t BUF = (size_t)BATCHN * SEQL * DMODEL * sizeof(float);  // 16.78 MB
  float*    Qb = (float*)(ws + 0 * BUF);
  float*    Kb = (float*)(ws + 1 * BUF);
  float*    Vb = (float*)(ws + 2 * BUF);
  float*    Cb = (float*)(ws + 3 * BUF);
  float*    Ab = Qb;                         // attn_out overwrites Q (dead)
  float*    X1 = Kb;                         // LN1 output overwrites K (dead)
  _Float16* Hb = (_Float16*)(ws + 2 * BUF);  // FFN hidden (f16) over V+CTX (dead)
  float*    Fb = Qb;                         // FFN output overwrites attn_out (dead)
  float*    Ob = (float*)d_out;

  const int M = BATCHN * SEQL;  // 8192
  dim3 blk(256);
  dim3 gp(DMODEL / 64, M / 128);

  wmma_gemm<float, float, false><<<gp, blk, 0, stream>>>(x, Wq, bq, Qb, M, DMODEL, DMODEL);
  wmma_gemm<float, float, false><<<gp, blk, 0, stream>>>(x, Wk, bk, Kb, M, DMODEL, DMODEL);
  wmma_gemm<float, float, false><<<gp, blk, 0, stream>>>(x, Wv, bv, Vb, M, DMODEL, DMODEL);

  dim3 gf(SEQL / 64, NHEADS, BATCHN);
  flash_attn<<<gf, blk, 0, stream>>>(Qb, Kb, Vb, rb, Cb);

  wmma_gemm<float, float, false><<<gp, blk, 0, stream>>>(Cb, Wo, bo, Ab, M, DMODEL, DMODEL);
  add_ln<<<M, 256, 0, stream>>>(x, Ab, l1s, l1b, X1);

  dim3 g1(DFF / 64, M / 128);
  wmma_gemm<float, _Float16, true><<<g1, blk, 0, stream>>>(X1, W1, b1, Hb, M, DFF, DMODEL);
  dim3 g2(DMODEL / 64, M / 128);
  wmma_gemm<_Float16, float, false><<<g2, blk, 0, stream>>>(Hb, W2, b2, Fb, M, DMODEL, DFF);

  add_ln<<<M, 256, 0, stream>>>(X1, Fb, l2s, l2b, Ob);
}
